// ManhattanLoss_77764677861984
// MI455X (gfx1250) — matrix operand order repaired and checked by reading, not compiled, not run on hardware
//
#include <hip/hip_runtime.h>

// ---------- types ----------
typedef _Float16 h2   __attribute__((ext_vector_type(2)));
typedef _Float16 v16h __attribute__((ext_vector_type(16)));
typedef float    v8f  __attribute__((ext_vector_type(8)));
typedef unsigned int u32x8 __attribute__((ext_vector_type(8)));

#define NROW 4096
#define NDIM 256
#define NDP  128          // NDIM/2 packed f16 pairs
#define NPOS 32768
#define TEMP 0.07f
#define JSPLIT 4          // j-range splits per i-block (grid = 64*JSPLIT)

// |a-b| on packed f16 pair: v_pk_add_f16(neg) + v_and_b32
__device__ __forceinline__ unsigned int absdiff2(unsigned int a, unsigned int b) {
    h2 x = __builtin_bit_cast(h2, a);
    h2 y = __builtin_bit_cast(h2, b);
    h2 d = x - y;
    return __builtin_bit_cast(unsigned int, d) & 0x7FFF7FFFu;
}

// Build one 16x32 f16 A-fragment (abs-diffs) for d-pair index dp.
// A layout (contiguous k=16*hi+s, as B): lane holds row i=L&15; VGPR v<4 -> j = v+8*hi,
// v>=4 -> j = 4+(v-4)+8*hi ; the two f16 halves are (d0,d1) of the pair.
__device__ __forceinline__ v16h buildA(const unsigned int* ldsI,
                                       const unsigned int* ldsJ,
                                       int dp, int ibase, int jA, int jB) {
    const unsigned int xi = ldsI[dp * 64 + ibase];
    const uint4 a = *(const uint4*)(ldsJ + dp * 64 + jA);
    const uint4 b = *(const uint4*)(ldsJ + dp * 64 + jB);
    u32x8 aw;
    aw[0] = absdiff2(xi, a.x);
    aw[1] = absdiff2(xi, a.y);
    aw[2] = absdiff2(xi, a.z);
    aw[3] = absdiff2(xi, a.w);
    aw[4] = absdiff2(xi, b.x);
    aw[5] = absdiff2(xi, b.y);
    aw[6] = absdiff2(xi, b.z);
    aw[7] = absdiff2(xi, b.w);
    return __builtin_bit_cast(v16h, aw);
}

#define WMMA(Cacc, Afrag) \
    __builtin_amdgcn_wmma_f32_16x16x32_f16(false, (Afrag), false, B, (short)0, (Cacc), false, false)

// ---------- kernel 1: pack f32 -> packed f16 pairs, d-pair-major ----------
// xP[dp][j] = (f16(x[j][2dp]), f16(x[j][2dp+1])), dp in [0,128), j in [0,4096)
__global__ __launch_bounds__(256) void pack_kernel(const float* __restrict__ x,
                                                   unsigned int* __restrict__ xP) {
    const int dp = blockIdx.x >> 4;                        // 0..127
    const int j  = ((blockIdx.x & 15) << 8) + threadIdx.x; // 0..4095
    const float2 v = ((const float2*)x)[j * NDP + dp];
    h2 hv;
    hv[0] = (_Float16)v.x;
    hv[1] = (_Float16)v.y;
    xP[dp * NROW + j] = __builtin_bit_cast(unsigned int, hv);
}

// ---------- kernel 2: WMMA L1-distance tiles + exp-sum per row ----------
// grid.x = 64 * JSPLIT ; block = 512 (16 waves)
// wave w: i-tile = w&3 (16 rows), j-group = w>>2 (16 cols within 64-col panel)
__global__ __launch_bounds__(512) void l1_expsum_kernel(const unsigned int* __restrict__ xP,
                                                        float* __restrict__ sbuf) {
    __shared__ __align__(16) unsigned int ldsJ[NDP * 64];  // 32 KB j-panel
    __shared__ __align__(16) unsigned int ldsI[NDP * 64];  // 32 KB i-panel

    const int tid  = threadIdx.x;
    const int lane = tid & 31;
    const int wave = tid >> 5;
    const int n16  = lane & 15;
    const int hi   = lane >> 4;          // lane half
    const int it   = wave & 3;           // i-tile in block
    const int jg   = wave >> 2;          // j-tile in 64-col panel
    const int bi   = blockIdx.x & 63;
    const int js   = blockIdx.x >> 6;    // j-split 0..JSPLIT-1
    const int i0   = bi * 64;

    // stage I-panel: columns i0..i0+63 of xP  (8192 u32 / 512 thr = 4x uint4)
    {
        const int c4  = (tid & 15) * 4;
        const int dpr = tid >> 4;
        #pragma unroll
        for (int r = 0; r < 4; ++r) {
            const int dp = dpr + 32 * r;
            *(uint4*)(&ldsI[dp * 64 + c4]) =
                *(const uint4*)(xP + dp * NROW + i0 + c4);
        }
    }

    // constant selection matrix B[k][n] = 1 iff k in {2n, 2n+1}
    // B layout: lane=column n (L&15); lanes<16 hold K=0..15 (VGPR v: K=2v,2v+1),
    //           lanes>=16 hold K=16..31 (VGPR v: K=16+2v,16+2v+1)
    u32x8 bw;
    #pragma unroll
    for (int v = 0; v < 8; ++v) {
        const int  sel = hi ? (n16 - 8) : n16;
        const bool on  = (hi ? (n16 >= 8) : (n16 < 8)) && (sel == v);
        bw[v] = on ? 0x3C003C00u : 0u;    // (1.0h, 1.0h)
    }
    const v16h B = __builtin_bit_cast(v16h, bw);

    float acc[8] = {0.f,0.f,0.f,0.f,0.f,0.f,0.f,0.f};
    const int jbase = jg * 16;
    const int jAo   = jbase + hi * 8;
    const int jBo   = jbase + hi * 8 + 4;
    const int ibase = it * 16 + n16;

    for (int jb = js * (NROW / 64 / JSPLIT); jb < (js + 1) * (NROW / 64 / JSPLIT); ++jb) {
        __syncthreads();
        {   // stage 64-column j-panel
            const int c4  = (tid & 15) * 4;
            const int dpr = tid >> 4;
            #pragma unroll
            for (int r = 0; r < 4; ++r) {
                const int dp = dpr + 32 * r;
                *(uint4*)(&ldsJ[dp * 64 + c4]) =
                    *(const uint4*)(xP + dp * NROW + jb * 64 + c4);
            }
        }
        __syncthreads();

        // 4-buffer rotating software pipeline: every A-buffer rebuild sits
        // >=16 independent VALU after the WMMA that last read it -> no WAR nops.
        // Two accumulator chains keep consecutive WMMAs independent.
        v8f c0 = {}, c1 = {};
        v16h A0 = buildA(ldsI, ldsJ, 0, ibase, jAo, jBo);
        v16h A1 = buildA(ldsI, ldsJ, 1, ibase, jAo, jBo);
        v16h A2 = buildA(ldsI, ldsJ, 2, ibase, jAo, jBo);
        v16h A3;
        int dp = 0;
        #pragma unroll 1
        for (; dp <= NDP - 8; dp += 4) {
            c0 = WMMA(c0, A0);
            A3 = buildA(ldsI, ldsJ, dp + 3, ibase, jAo, jBo);
            c1 = WMMA(c1, A1);
            A0 = buildA(ldsI, ldsJ, dp + 4, ibase, jAo, jBo);
            c0 = WMMA(c0, A2);
            A1 = buildA(ldsI, ldsJ, dp + 5, ibase, jAo, jBo);
            c1 = WMMA(c1, A3);
            A2 = buildA(ldsI, ldsJ, dp + 6, ibase, jAo, jBo);
        }
        // epilogue: dp==124 here; A0=b(124), A1=b(125), A2=b(126)
        c0 = WMMA(c0, A0);
        A3 = buildA(ldsI, ldsJ, NDP - 1, ibase, jAo, jBo);
        c1 = WMMA(c1, A1);
        c0 = WMMA(c0, A2);
        c1 = WMMA(c1, A3);

        // sim = -D/T, logits_max == 0 by construction; mask diagonal; sum exp
        const int gj = jb * 64 + jbase + n16;
        #pragma unroll
        for (int r = 0; r < 8; ++r) {
            const int gi = i0 + it * 16 + r + 8 * hi;   // C/D layout: M=r+8*hi, N=n16
            const float d = c0[r] + c1[r];
            float e = __expf(d * (-1.0f / TEMP));
            if (gi == gj) e = 0.0f;
            acc[r] += e;
        }
    }

    // reduce each row over the 16 lanes of its half (xor 1,2,4,8 stay in-half)
    #pragma unroll
    for (int r = 0; r < 8; ++r) {
        float v = acc[r];
        v += __shfl_xor(v, 1, 32);
        v += __shfl_xor(v, 2, 32);
        v += __shfl_xor(v, 4, 32);
        v += __shfl_xor(v, 8, 32);
        acc[r] = v;
    }

    __syncthreads();                    // ldsJ dead -> reuse as partials[64][4]
    float* partials = (float*)ldsJ;
    if (n16 == 0) {
        #pragma unroll
        for (int r = 0; r < 8; ++r)
            partials[(it * 16 + r + 8 * hi) * 4 + jg] = acc[r];
    }
    __syncthreads();
    if (tid < 64) {
        const float s = partials[tid * 4] + partials[tid * 4 + 1] +
                        partials[tid * 4 + 2] + partials[tid * 4 + 3];
        sbuf[js * NROW + i0 + tid] = s;  // deterministic: unique (js,row) writer
    }
}

// ---------- kernel 3: positive pairs, exact f32, per-block partial sums ----------
__global__ __launch_bounds__(256) void pairs_kernel(const float* __restrict__ x,
                                                    const int* __restrict__ row,
                                                    const int* __restrict__ col,
                                                    const float* __restrict__ sbuf,
                                                    float* __restrict__ pbuf) {
    const int k  = blockIdx.x * 256 + threadIdx.x;
    const int r  = row[k];
    const int cl = col[k];
    const float4* xr = (const float4*)(x + r * NDIM);
    const float4* xc = (const float4*)(x + cl * NDIM);
    float s = 0.0f;
    #pragma unroll 8
    for (int t = 0; t < NDIM / 4; ++t) {
        const float4 a = xr[t];
        const float4 b = xc[t];
        s += fabsf(a.x - b.x) + fabsf(a.y - b.y) + fabsf(a.z - b.z) + fabsf(a.w - b.w);
    }
    const float denomsum = sbuf[r] + sbuf[NROW + r] + sbuf[2 * NROW + r] + sbuf[3 * NROW + r];
    float lp = s * (-1.0f / TEMP) - __logf(denomsum);   // log(0) -> -inf, faithful
    lp += __shfl_xor(lp, 1, 32);
    lp += __shfl_xor(lp, 2, 32);
    lp += __shfl_xor(lp, 4, 32);
    lp += __shfl_xor(lp, 8, 32);
    lp += __shfl_xor(lp, 16, 32);
    __shared__ float wsum[8];
    if ((threadIdx.x & 31) == 0) wsum[threadIdx.x >> 5] = lp;
    __syncthreads();
    if (threadIdx.x == 0) {
        float b = 0.0f;
        #pragma unroll
        for (int i = 0; i < 8; ++i) b += wsum[i];
        pbuf[blockIdx.x] = b;            // deterministic per-block partial
    }
}

// ---------- kernel 4: deterministic final reduce ----------
__global__ __launch_bounds__(128) void final_kernel(const float* __restrict__ pbuf,
                                                    float* __restrict__ out) {
    const int t = threadIdx.x;           // 128 threads
    float v = pbuf[t];
    v += __shfl_xor(v, 1, 32);
    v += __shfl_xor(v, 2, 32);
    v += __shfl_xor(v, 4, 32);
    v += __shfl_xor(v, 8, 32);
    v += __shfl_xor(v, 16, 32);
    __shared__ float w[4];
    if ((t & 31) == 0) w[t >> 5] = v;
    __syncthreads();
    if (t == 0)
        out[0] = -TEMP * (w[0] + w[1] + w[2] + w[3]) * (1.0f / (float)NPOS);
}

extern "C" void kernel_launch(void* const* d_in, const int* in_sizes, int n_in,
                              void* d_out, int out_size, void* d_ws, size_t ws_size,
                              hipStream_t stream) {
    (void)in_sizes; (void)n_in; (void)out_size; (void)ws_size;
    const float* x   = (const float*)d_in[0];
    const int*   row = (const int*)d_in[1];
    const int*   col = (const int*)d_in[2];
    float* out = (float*)d_out;

    unsigned char* ws = (unsigned char*)d_ws;
    unsigned int* xP = (unsigned int*)ws;                                    // 2 MB
    float* sbuf = (float*)(ws + (size_t)NDP * NROW * 4);                     // 64 KB
    float* pbuf = (float*)(ws + (size_t)NDP * NROW * 4 + JSPLIT * NROW * 4); // 512 B

    pack_kernel<<<2048, 256, 0, stream>>>(x, xP);
    l1_expsum_kernel<<<64 * JSPLIT, 512, 0, stream>>>(xP, sbuf);
    pairs_kernel<<<NPOS / 256, 256, 0, stream>>>(x, row, col, sbuf, pbuf);
    final_kernel<<<1, 128, 0, stream>>>(pbuf, out);
}
